// CostVolume_31980326486033
// MI455X (gfx1250) — matrix operand order repaired and checked by reading, not compiled, not run on hardware
//
#include <hip/hip_runtime.h>
#include <hip/hip_bf16.h>

// ---------------------------------------------------------------------------
// Stereo cost-volume pipeline for MI455X (gfx1250, wave32, WMMA).
//   down:   1x1 conv (1024->64) + BN + ReLU for L and R  -> bf16, pixel-major
//   conv3a: 3x3x3 conv (128->64) over implicit cost volume (built on the fly
//           in LDS from L/R with the disparity shift+mask) + BN + ReLU
//   conv3b: 3x3x3 conv (64->64) + BN + ReLU -> fp32 output [CF*D, H, W]
// GEMMs: v_wmma_f32_16x16x32_bf16 (fp32 accumulate).
// Tile staging: TENSOR_LOAD_TO_LDS (TDM) for interior conv3b tiles,
//               global_load_async_to_lds_b128 + s_wait_asynccnt elsewhere.
// ---------------------------------------------------------------------------

#define HH_ 48
#define WW_ 160
#define CIN_ 1024
#define CF_ 64
#define DD_ 48
#define NPIX_ (HH_ * WW_)      // 7680

typedef __bf16 bf16x16 __attribute__((ext_vector_type(16)));
typedef float  f32x8   __attribute__((ext_vector_type(8)));
typedef unsigned short u16x8 __attribute__((ext_vector_type(8)));
typedef unsigned int   uint4v __attribute__((ext_vector_type(4)));
typedef int            int8v  __attribute__((ext_vector_type(8)));
typedef int            int4v  __attribute__((ext_vector_type(4)));

union FragU { bf16x16 v; u16x8 h[2]; };

__device__ __forceinline__ unsigned short f2bf(float f) {
    unsigned int u = __float_as_uint(f);
    u += 0x7FFFu + ((u >> 16) & 1u);            // round-to-nearest-even
    return (unsigned short)(u >> 16);
}

// ---------------- workspace layout (bytes, 256-aligned) --------------------
// needs ~48 MB of d_ws
static const size_t OFF_W1  = 0;                       // 64*1024   bf16
static const size_t OFF_WA  = 131072;                  // 27*64*128 bf16
static const size_t OFF_WB  = 573440;                  // 27*64*64  bf16
static const size_t OFF_S1  = 794624;                  // 6 x 64 f32 params
static const size_t OFF_Q1  = OFF_S1 + 256;
static const size_t OFF_S3A = OFF_S1 + 512;
static const size_t OFF_Q3A = OFF_S1 + 768;
static const size_t OFF_S3B = OFF_S1 + 1024;
static const size_t OFF_Q3B = OFF_S1 + 1280;
static const size_t OFF_ZERO = 796160;                 // 4096 B zero page
static const size_t OFF_LBT = 800256;                  // 7680*64 bf16 (pixel-major)
static const size_t OFF_RBT = OFF_LBT + 983040;
static const size_t OFF_Y1  = OFF_RBT + 983040;        // 368640*64 bf16 (pixel-major)

// ---------------------------------------------------------------------------
// prep: bf16 weight re-layout ([tap][o][i], i contiguous) + BN folding +
//       zero page for OOB async-copy sources.
// ---------------------------------------------------------------------------
__global__ void prep_kernel(
    const float* __restrict__ c3a_w, const float* __restrict__ c3b_w,
    const float* __restrict__ conv1_w,
    const float* __restrict__ c1b, const float* __restrict__ g1,
    const float* __restrict__ b1,  const float* __restrict__ m1, const float* __restrict__ v1,
    const float* __restrict__ c3ab, const float* __restrict__ g3a,
    const float* __restrict__ b3a,  const float* __restrict__ m3a, const float* __restrict__ v3a,
    const float* __restrict__ c3bb, const float* __restrict__ g3b,
    const float* __restrict__ b3b,  const float* __restrict__ m3b, const float* __restrict__ v3b,
    unsigned short* __restrict__ Wa, unsigned short* __restrict__ Wb,
    unsigned short* __restrict__ w1b,
    float* __restrict__ s1, float* __restrict__ q1,
    float* __restrict__ s3a, float* __restrict__ q3a,
    float* __restrict__ s3b, float* __restrict__ q3b,
    float* __restrict__ zp)
{
    const int NA = 27 * 64 * 128;       // Wa elements
    const int NB = 27 * 64 * 64;        // Wb elements
    const int NC = 64 * 1024;           // w1b elements
    const int total = NA + NB + NC + 64 + 1024;
    for (int k = blockIdx.x * blockDim.x + threadIdx.x; k < total;
         k += gridDim.x * blockDim.x) {
        if (k < NA) {
            // Wa[(tap*64+o)*128+i] = c3a_w[o][i][tap]   (OIDHW, D/H/W flatten to tap)
            int i = k & 127, o = (k >> 7) & 63, tap = k >> 13;
            Wa[k] = f2bf(c3a_w[(o * 128 + i) * 27 + tap]);
        } else if (k < NA + NB) {
            int kk = k - NA;
            int i = kk & 63, o = (kk >> 6) & 63, tap = kk >> 12;
            Wb[kk] = f2bf(c3b_w[(o * 64 + i) * 27 + tap]);
        } else if (k < NA + NB + NC) {
            int kk = k - NA - NB;
            w1b[kk] = f2bf(conv1_w[kk]);
        } else if (k < NA + NB + NC + 64) {
            int o = k - NA - NB - NC;
            float s;
            s = g1[o] * rsqrtf(v1[o] + 1e-5f);
            s1[o] = s;  q1[o] = (c1b[o] - m1[o]) * s + b1[o];
            s = g3a[o] * rsqrtf(v3a[o] + 1e-5f);
            s3a[o] = s; q3a[o] = (c3ab[o] - m3a[o]) * s + b3a[o];
            s = g3b[o] * rsqrtf(v3b[o] + 1e-5f);
            s3b[o] = s; q3b[o] = (c3bb[o] - m3b[o]) * s + b3b[o];
        } else {
            zp[k - NA - NB - NC - 64] = 0.f;     // zero page
        }
    }
}

// ---------------------------------------------------------------------------
// down: y[o,p] = relu(bn(sum_i w1[o,i]*x[i,p] + bias))  -> bf16 [p][o]
// GEMM: A = weights (M=o), B = activations via LDS transpose (fp32->bf16,
// must convert in flight, so VALU staging, not async).
// One workgroup = 128 pixels; 8 waves, each wave owns 16 pixels x 64 outch.
// ---------------------------------------------------------------------------
__global__ void __launch_bounds__(256) down_kernel(
    const float* __restrict__ xL, const float* __restrict__ xR,
    const unsigned short* __restrict__ w1b,
    const float* __restrict__ s1, const float* __restrict__ q1,
    unsigned short* __restrict__ LbT, unsigned short* __restrict__ RbT)
{
    __shared__ __attribute__((aligned(16))) unsigned short xT[128 * 48]; // [pix][k], padded row 48

    const float* x = blockIdx.y ? xR : xL;
    unsigned short* outT = blockIdx.y ? RbT : LbT;
    const int t = threadIdx.x;
    const int lane = t & 31, wid = t >> 5;
    const int p0 = blockIdx.x * 128;

    f32x8 acc[4];
    #pragma unroll
    for (int f = 0; f < 4; ++f)
        for (int j = 0; j < 8; ++j) acc[f][j] = 0.f;

    for (int i0 = 0; i0 < CIN_; i0 += 32) {
        __syncthreads();
        // stage 32 channels x 128 pixels, fp32 -> bf16, transposed to [pix][k]
        #pragma unroll
        for (int r = 0; r < 16; ++r) {
            int e = r * 256 + t;
            int pi = e & 127, ci = e >> 7;
            xT[pi * 48 + ci] = f2bf(x[(i0 + ci) * NPIX_ + p0 + pi]);
        }
        __syncthreads();

        FragU b;   // B: column n = lane%16 (pixel), K chunks at 8*(lane/16) and +16
        const unsigned short* pb = &xT[(wid * 16 + (lane & 15)) * 48 + 8 * (lane >> 4)];
        b.h[0] = *(const u16x8*)pb;
        b.h[1] = *(const u16x8*)(pb + 16);
        #pragma unroll
        for (int f = 0; f < 4; ++f) {
            FragU a;  // A: row m = lane%16 (outch), K chunks at 8*(lane/16) and +16
            const unsigned short* pa =
                w1b + (f * 16 + (lane & 15)) * CIN_ + i0 + 8 * (lane >> 4);
            a.h[0] = *(const u16x8*)pa;
            a.h[1] = *(const u16x8*)(pa + 16);
            acc[f] = __builtin_amdgcn_wmma_f32_16x16x32_bf16(
                false, a.v, false, b.v, (short)0, acc[f], false, false);
        }
    }

    // epilogue: BN + ReLU, store pixel-major bf16
    const int p = p0 + wid * 16 + (lane & 15);
    #pragma unroll
    for (int f = 0; f < 4; ++f)
        #pragma unroll
        for (int j = 0; j < 8; ++j) {
            int o = f * 16 + j + 8 * (lane >> 4);   // C layout: M = j + 8*(lane/16)
            float y = acc[f][j] * s1[o] + q1[o];
            outT[p * 64 + o] = f2bf(fmaxf(y, 0.f));
        }
}

// ---------------------------------------------------------------------------
// conv3: 3x3x3 conv as 27 taps of 1x1 GEMMs, implicit-GEMM via LDS tile.
// IS_A=true : input = on-the-fly cost volume from LbT/RbT (shift + mask)
// IS_A=false: input = y1T (conv3a output, bf16 pixel-major)
// Workgroup: fixed (d,h), 64-wide W tile; 8 waves = 4 N-frags x 2 M-pairs,
// each wave computes 2 M-frags so every B fragment feeds 2 WMMAs.
// Staging: interior conv3b slabs -> one TDM TENSOR_LOAD_TO_LDS (wave 0,
// descriptor per ISA 8.3-8.6, s_wait_tensorcnt); everything else ->
// GLOBAL_LOAD_ASYNC_TO_LDS_B128 with OOB redirected to a zero page and tail
// lanes to a dummy LDS slot so EXEC is all-ones for every async issue.
// ---------------------------------------------------------------------------
template <int IC, bool IS_A>
__global__ void __launch_bounds__(256) conv3_kernel(
    const unsigned short* __restrict__ srcL,
    const unsigned short* __restrict__ srcR,
    const unsigned short* __restrict__ zp,     // 4 KB zero page
    const unsigned short* __restrict__ Wt,     // [27][64][IC] bf16
    const float* __restrict__ sc, const float* __restrict__ sh,
    unsigned short* __restrict__ outBf,        // conv3a -> y1T
    float* __restrict__ outF)                  // conv3b -> final
{
    constexpr int TW = 66;                               // 64 + 2 halo
    __shared__ __attribute__((aligned(16))) unsigned short tile[3 * TW * IC + 8];

    const int t = threadIdx.x, lane = t & 31, wid = t >> 5;
    const int nfrag = wid & 3, mpair = wid >> 2;
    const int w0 = blockIdx.x * 64, h = blockIdx.y, d = blockIdx.z;

    // pull the first weight fragment line toward L2/L0 (global_prefetch_b8)
    __builtin_prefetch(Wt + ((size_t)nfrag * 16 + (lane & 15)) * IC, 0, 1);

    f32x8 acc[2];
    #pragma unroll
    for (int mi = 0; mi < 2; ++mi)
        for (int j = 0; j < 8; ++j) acc[mi][j] = 0.f;

    constexpr int NBLK = 3 * TW * (IC / 8);              // 16B blocks per kd slab
    constexpr int ITER = (NBLK + 255) / 256;

    for (int kd = 0; kd < 3; ++kd) {
        const int dd = d + kd - 1;
        __syncthreads();                                  // tile reuse guard

        // block-uniform: full 3x66xIC slab in-bounds, unmasked? (conv3b only)
        bool tdm_ok = false;
#if __has_builtin(__builtin_amdgcn_tensor_load_to_lds)
        if (!IS_A)
            tdm_ok = (blockIdx.x == 1) && (h >= 1) && (h <= HH_ - 2) &&
                     (dd >= 0) && (dd < DD_);
#endif
        if (tdm_ok) {
#if __has_builtin(__builtin_amdgcn_tensor_load_to_lds)
            if (wid == 0) {
                // D#: 3D tile, x = channel (64, contiguous), y = w (66 lines,
                // stride 64 elems), z = h (3 planes, stride 160*64 elems).
                unsigned long long ga = (unsigned long long)(size_t)
                    &srcL[((size_t)(dd * HH_ + (h - 1)) * WW_ + (w0 - 1)) * 64];
                unsigned ldsa = (unsigned)(size_t)&tile[0];   // low 32b = LDS off
                uint4v g0;
                g0[0] = 1u;                                    // count=1 (valid)
                g0[1] = ldsa;                                  // lds_addr
                g0[2] = (unsigned)ga;                          // global_addr[31:0]
                g0[3] = (unsigned)((ga >> 32) & 0x01FFFFFFu) | 0x80000000u; // [56:32]|type=2
                int8v g1;
                g1[0] = 1 << 16;                               // data_size = 2B
                g1[1] = 64 << 16;                              // tensor_dim0 = 64
                g1[2] = 66 << 16;                              // tensor_dim1 = 66
                g1[3] = 64 << 16;                              // tile_dim0 = 64
                g1[4] = 66 | (3 << 16);                        // tile_dim1=66, tile_dim2=3
                g1[5] = 64;                                    // tensor_dim0_stride
                g1[6] = (int)(10240u << 16);                   // tensor_dim1_stride = 160*64
                g1[7] = 0;
                int4v g2; g2[0] = 3; g2[1] = 0; g2[2] = 0; g2[3] = 0; // tensor_dim2=3
                int4v g3; g3[0] = 0; g3[1] = 0; g3[2] = 0; g3[3] = 0;
                int8v g4;                                      // reserved/extended: zeros
                for (int z = 0; z < 8; ++z) g4[z] = 0;
                __builtin_amdgcn_tensor_load_to_lds(g0, g1, g2, g3, g4, 0);
            }
            __builtin_amdgcn_s_wait_tensorcnt(0);
#endif
        } else {
            for (int it = 0; it < ITER; ++it) {
                int blk = it * 256 + t;
                bool live = blk < NBLK;
                int cb  = blk % (IC / 8);
                int wpr = (blk / (IC / 8)) % TW;
                int p3  = blk / ((IC / 8) * TW);
                int hh = h + p3 - 1;
                int ww = w0 - 1 + wpr;
                int c0 = cb * 8;
                bool inb = live && (dd >= 0) && (dd < DD_) && (hh >= 0) &&
                           (hh < HH_) && (ww >= 0) && (ww < WW_);
                const unsigned short* src = zp;           // OOB / masked -> zeros
                if (IS_A) {
                    // cost volume: c<64 -> L[h][w] if w>=d; c>=64 -> R[h][w-d] if w>=d
                    if (inb && ww >= dd)
                        src = (c0 < 64)
                            ? &srcL[(hh * WW_ + ww) * 64 + c0]
                            : &srcR[(hh * WW_ + (ww - dd)) * 64 + (c0 - 64)];
                } else {
                    if (inb) src = &srcL[((dd * HH_ + hh) * WW_ + ww) * 64 + c0];
                }
                const unsigned short* dst = live ? &tile[(p3 * TW + wpr) * IC + c0]
                                                 : &tile[3 * TW * IC];
                unsigned ldsoff = (unsigned)(size_t)dst;  // low 32b = LDS offset
                asm volatile("global_load_async_to_lds_b128 %0, %1, off"
                             :: "v"(ldsoff), "v"(src) : "memory");
            }
            asm volatile("s_wait_asynccnt 0" ::: "memory");
        }
        __syncthreads();

        for (int kh = 0; kh < 3; ++kh)
            for (int kw = 0; kw < 3; ++kw) {
                const int tap = (kd * 3 + kh) * 3 + kw;
                #pragma unroll
                for (int ic0 = 0; ic0 < IC; ic0 += 32) {
                    FragU b;  // B: col n = lane%16 (outch), K chunks 8*(lane/16), +16
                    const unsigned short* pb =
                        Wt + ((size_t)tap * 64 + nfrag * 16 + (lane & 15)) * IC +
                        ic0 + 8 * (lane >> 4);
                    b.h[0] = *(const u16x8*)pb;
                    b.h[1] = *(const u16x8*)(pb + 16);
                    #pragma unroll
                    for (int mi = 0; mi < 2; ++mi) {
                        FragU a;  // A: row m = lane%16 (pixel), K chunks 8*(lane/16), +16
                        const int wloc = (mpair * 2 + mi) * 16 + (lane & 15) + kw;
                        const unsigned short* pa =
                            &tile[(kh * TW + wloc) * IC + ic0 + 8 * (lane >> 4)];
                        a.h[0] = *(const u16x8*)pa;
                        a.h[1] = *(const u16x8*)(pa + 16);
                        acc[mi] = __builtin_amdgcn_wmma_f32_16x16x32_bf16(
                            false, a.v, false, b.v, (short)0, acc[mi], false, false);
                    }
                }
            }
    }

    // epilogue: BN + ReLU (partial right tile: guard w)
    const int o = nfrag * 16 + (lane & 15);
    const float s = sc[o], q = sh[o];
    #pragma unroll
    for (int mi = 0; mi < 2; ++mi)
        #pragma unroll
        for (int j = 0; j < 8; ++j) {
            int w = w0 + (mpair * 2 + mi) * 16 + j + 8 * (lane >> 4);
            if (w < WW_) {
                float y = fmaxf(acc[mi][j] * s + q, 0.f);
                if (IS_A)
                    outBf[((size_t)(d * HH_ + h) * WW_ + w) * 64 + o] = f2bf(y);
                else
                    outF[(((size_t)o * DD_ + d) * HH_ + h) * WW_ + w] = y;
            }
        }
}

// ---------------------------------------------------------------------------
extern "C" void kernel_launch(void* const* d_in, const int* in_sizes, int n_in,
                              void* d_out, int out_size, void* d_ws, size_t ws_size,
                              hipStream_t stream)
{
    const float* xL      = (const float*)d_in[0];
    const float* xR      = (const float*)d_in[1];
    const float* conv1_w = (const float*)d_in[2];
    const float* conv1_b = (const float*)d_in[3];
    const float* g1 = (const float*)d_in[4],  *b1 = (const float*)d_in[5];
    const float* m1 = (const float*)d_in[6],  *v1 = (const float*)d_in[7];
    const float* c3a_w = (const float*)d_in[8], *c3a_b = (const float*)d_in[9];
    const float* g3a = (const float*)d_in[10], *b3a = (const float*)d_in[11];
    const float* m3a = (const float*)d_in[12], *v3a = (const float*)d_in[13];
    const float* c3b_w = (const float*)d_in[14], *c3b_b = (const float*)d_in[15];
    const float* g3b = (const float*)d_in[16], *b3b = (const float*)d_in[17];
    const float* m3b = (const float*)d_in[18], *v3b = (const float*)d_in[19];

    char* ws = (char*)d_ws;
    unsigned short* w1b = (unsigned short*)(ws + OFF_W1);
    unsigned short* Wa  = (unsigned short*)(ws + OFF_WA);
    unsigned short* Wb  = (unsigned short*)(ws + OFF_WB);
    float* s1  = (float*)(ws + OFF_S1);
    float* q1  = (float*)(ws + OFF_Q1);
    float* s3a = (float*)(ws + OFF_S3A);
    float* q3a = (float*)(ws + OFF_Q3A);
    float* s3b = (float*)(ws + OFF_S3B);
    float* q3b = (float*)(ws + OFF_Q3B);
    float* zp  = (float*)(ws + OFF_ZERO);
    unsigned short* LbT = (unsigned short*)(ws + OFF_LBT);
    unsigned short* RbT = (unsigned short*)(ws + OFF_RBT);
    unsigned short* y1T = (unsigned short*)(ws + OFF_Y1);

    prep_kernel<<<1024, 256, 0, stream>>>(
        c3a_w, c3b_w, conv1_w,
        conv1_b, g1, b1, m1, v1,
        c3a_b, g3a, b3a, m3a, v3a,
        c3b_b, g3b, b3b, m3b, v3b,
        Wa, Wb, w1b, s1, q1, s3a, q3a, s3b, q3b, zp);

    down_kernel<<<dim3(NPIX_ / 128, 2), 256, 0, stream>>>(
        xL, xR, w1b, s1, q1, LbT, RbT);

    conv3_kernel<128, true><<<dim3((WW_ + 63) / 64, HH_, DD_), 256, 0, stream>>>(
        LbT, RbT, (const unsigned short*)zp, Wa, s3a, q3a, y1T, nullptr);

    conv3_kernel<64, false><<<dim3((WW_ + 63) / 64, HH_, DD_), 256, 0, stream>>>(
        y1T, nullptr, (const unsigned short*)zp, Wb, s3b, q3b, nullptr, (float*)d_out);
}
